// LSA_47313359733211
// MI455X (gfx1250) — matrix operand order repaired and checked by reading, not compiled, not run on hardware
//
#include <hip/hip_runtime.h>
#include <hip/hip_bf16.h>

// ---------------------------------------------------------------------------
// LSA attention for MI455X (gfx1250): bf16 WMMA everywhere, fp32 accumulate.
// ---------------------------------------------------------------------------

typedef __attribute__((ext_vector_type(16))) __bf16 v16bf;
typedef __attribute__((ext_vector_type(8)))  __bf16 v8bf;
typedef __attribute__((ext_vector_type(8)))  float  v8f;

#define B_    4
#define S_    2048
#define DIM_  512
#define HEADS_ 8
#define DHEAD_ 64
#define INNER_ 512
#define NQKV_ 1536
#define MROWS_ (B_ * S_)     // 8192

__device__ __forceinline__ v8f wmma_bf16(v16bf a, v16bf b, v8f c) {
  return __builtin_amdgcn_wmma_f32_16x16x32_bf16(
      /*neg_a=*/false, a, /*neg_b=*/false, b,
      /*c_mod=*/(short)0, c, /*reuse_a=*/false, /*reuse_b=*/false);
}

__device__ __forceinline__ v16bf make_v16(v8bf lo, v8bf hi) {
  v16bf r;
#pragma unroll
  for (int i = 0; i < 8; ++i) { r[i] = lo[i]; r[i + 8] = hi[i]; }
  return r;
}

// A fragment (16x32 bf16, MxK): lane holds row = lane%16; per-lane elements are
// 16 contiguous K-values: [k0 + half*16 .. +15].
__device__ __forceinline__ v16bf load_afrag(const __bf16* base, int pitch,
                                            int row, int k0, int half) {
  const __bf16* p = base + (long)row * pitch + k0 + half * 16;
  v8bf lo = *(const v8bf*)(p);
  v8bf hi = *(const v8bf*)(p + 8);
  return make_v16(lo, hi);
}

// B fragment (32x16 bf16, KxN) from K-contiguous (transposed) storage:
// lane holds column n = lane%16; elements are 16 contiguous K at k0 + half*16.
__device__ __forceinline__ v16bf load_bfrag(const __bf16* bt, int pitch,
                                            int n, int k0, int half) {
  return *(const v16bf*)(bt + (long)n * pitch + k0 + half * 16);
}

// ---------------------------------------------------------------------------
// Kernel 0: fp32 -> bf16 convert (x -> Xb workspace)
// ---------------------------------------------------------------------------
__global__ __launch_bounds__(256) void lsa_cvt(const float* __restrict__ x,
                                               __bf16* __restrict__ y, int n) {
  int i = blockIdx.x * 256 + threadIdx.x;
  if (i < n) y[i] = (__bf16)x[i];
}

// ---------------------------------------------------------------------------
// Kernel 1: QKV projection. C[8192x1536] = Xb @ Wqkv, scattered to
// Q/K/V [bh][s][d] bf16, Q scaled by exp(log_temp).
// Block: 256 threads (8 waves), tile BM=128 x BN=64, full-K weight panel in LDS.
// ---------------------------------------------------------------------------
__global__ __launch_bounds__(256) void lsa_qkv_gemm(
    const __bf16* __restrict__ X,   // [8192][512]
    const float*  __restrict__ W,   // [512][1536]
    const float*  __restrict__ logt,
    __bf16* __restrict__ Q, __bf16* __restrict__ K, __bf16* __restrict__ V) {
  __shared__ __bf16 Bt[64][512];    // 64 KB, Bt[n][k]
  const int n0 = blockIdx.x * 64;
  const int m0 = blockIdx.y * 128;
  const int tid = threadIdx.x;

  for (int i = tid; i < 64 * 512; i += 256) {
    int n = i & 63, k = i >> 6;
    Bt[n][k] = (__bf16)W[(long)k * NQKV_ + n0 + n];
  }
  __syncthreads();

  const int wave = tid >> 5, lane = tid & 31;
  const int l16 = lane & 15, half = (lane >> 4) & 1;
  const int arow = m0 + wave * 16 + l16;

  v8f acc[4] = {};
  for (int ks = 0; ks < 16; ++ks) {
    v16bf a = load_afrag(X, DIM_, arow, ks * 32, half);
#pragma unroll
    for (int j = 0; j < 4; ++j) {
      v16bf bk = load_bfrag(&Bt[0][0], 512, j * 16 + l16, ks * 32, half);
      acc[j] = wmma_bf16(a, bk, acc[j]);
    }
  }

  const float temp = __expf(logt[0]);
#pragma unroll
  for (int j = 0; j < 4; ++j) {
#pragma unroll
    for (int r = 0; r < 8; ++r) {
      int m = m0 + wave * 16 + half * 8 + r;
      int n = n0 + j * 16 + l16;
      float vv = acc[j][r];
      int which = n >> 9;            // 0=q 1=k 2=v
      int rem = n & 511;
      int h = rem >> 6, d = rem & 63;
      int bb = m >> 11, s = m & 2047;
      long idx = (((long)(bb * HEADS_ + h) * S_) + s) * DHEAD_ + d;
      if (which == 0)      Q[idx] = (__bf16)(vv * temp);
      else if (which == 1) K[idx] = (__bf16)vv;
      else                 V[idx] = (__bf16)vv;
    }
  }
}

// ---------------------------------------------------------------------------
// Kernel 2: flash attention per (b,h). Block = 128 threads (4 waves),
// 64 query rows per block, streaming 64-wide K/V tiles. Diagonal masked.
// Output -> Aw [b][s][h*64+d] bf16.
// ---------------------------------------------------------------------------
__global__ __launch_bounds__(128) void lsa_attn(
    const __bf16* __restrict__ Q, const __bf16* __restrict__ K,
    const __bf16* __restrict__ V, __bf16* __restrict__ O) {
  __shared__ __bf16 Kt[64][64];       // K tile, [kv][d]  (row-major, d contiguous)
  __shared__ __bf16 Vt[64][64];       // V tile transposed, [d][kv]
  __shared__ __bf16 Ps[4][16][64];    // per-wave P strip, [row][kv]

  const int bh = blockIdx.y;
  const int q0 = blockIdx.x * 64;
  const int tid = threadIdx.x, wave = tid >> 5, lane = tid & 31;
  const int l16 = lane & 15, half = (lane >> 4) & 1;

  const __bf16* Qb = Q + (long)bh * S_ * DHEAD_;
  const __bf16* Kb = K + (long)bh * S_ * DHEAD_;
  const __bf16* Vb = V + (long)bh * S_ * DHEAD_;

  // Q fragments for this wave's 16 rows (resident across the whole kv loop)
  v16bf qf[2];
  {
    int row = q0 + wave * 16 + l16;
    qf[0] = load_afrag(Qb, DHEAD_, row, 0, half);
    qf[1] = load_afrag(Qb, DHEAD_, row, 32, half);
  }

  v8f acc[4] = {};
  float rmax[8], rsum[8];
#pragma unroll
  for (int r = 0; r < 8; ++r) { rmax[r] = -3.0e38f; rsum[r] = 0.f; }

  for (int kv0 = 0; kv0 < S_; kv0 += 64) {
    __syncthreads();   // previous iteration done with Kt/Vt
    // Stage K tile: 8KB as 16B chunks
    {
      const uint4* src = (const uint4*)(Kb + (long)kv0 * DHEAD_);
      uint4* dst = (uint4*)(&Kt[0][0]);
      for (int i = tid; i < 512; i += 128) dst[i] = src[i];
    }
    // Stage V tile transposed: Vt[d][kv]
    for (int i = tid; i < 512; i += 128) {
      int rr = i >> 3;            // kv row 0..63
      int c8 = (i & 7) * 8;       // d start
      const __bf16* sp = Vb + (long)(kv0 + rr) * DHEAD_ + c8;
#pragma unroll
      for (int e = 0; e < 8; ++e) Vt[c8 + e][rr] = sp[e];
    }
    if (kv0 + 64 < S_) {          // global_prefetch_b8 for next tiles
      __builtin_prefetch(Kb + (long)(kv0 + 64) * DHEAD_, 0, 0);
      __builtin_prefetch(Vb + (long)(kv0 + 64) * DHEAD_, 0, 0);
    }
    __syncthreads();

    // Scores S = Q @ K^T  (16 rows x 64 kv cols per wave)
    v8f sc[4] = {};
#pragma unroll
    for (int ks = 0; ks < 2; ++ks) {
#pragma unroll
      for (int j = 0; j < 4; ++j) {
        v16bf bk = *(const v16bf*)(&Kt[j * 16 + l16][ks * 32 + half * 16]);
        sc[j] = wmma_bf16(qf[ks], bk, sc[j]);
      }
    }

    // Diagonal mask + per-row tile max (rows live per-VGPR across 16-lane half)
    float tmax[8];
#pragma unroll
    for (int r = 0; r < 8; ++r) {
      int mrow = q0 + wave * 16 + half * 8 + r;
      float mx = -3.0e38f;
#pragma unroll
      for (int j = 0; j < 4; ++j) {
        int col = kv0 + j * 16 + l16;
        float sv = sc[j][r];
        if (col == mrow) { sv = -3.0e38f; sc[j][r] = sv; }
        mx = fmaxf(mx, sv);
      }
#pragma unroll
      for (int off = 1; off < 16; off <<= 1)
        mx = fmaxf(mx, __shfl_xor(mx, off, 16));
      tmax[r] = mx;
    }

    // Online softmax update; write P (bf16) to per-wave LDS strip
#pragma unroll
    for (int r = 0; r < 8; ++r) {
      float nm = fmaxf(rmax[r], tmax[r]);
      float sold = __expf(rmax[r] - nm);
      rmax[r] = nm;
      float tsum = 0.f;
#pragma unroll
      for (int j = 0; j < 4; ++j) {
        float p = __expf(sc[j][r] - nm);
        Ps[wave][half * 8 + r][j * 16 + l16] = (__bf16)p;
        tsum += p;
      }
#pragma unroll
      for (int off = 1; off < 16; off <<= 1)
        tsum += __shfl_xor(tsum, off, 16);
      rsum[r] = rsum[r] * sold + tsum;
#pragma unroll
      for (int j = 0; j < 4; ++j) acc[j][r] *= sold;
    }
    __syncthreads();   // ensure P strip stores are visible for re-layout reads

    // O += P @ V
#pragma unroll
    for (int ks = 0; ks < 2; ++ks) {
      const __bf16* pp = &Ps[wave][l16][ks * 32 + half * 16];
      v16bf ap = make_v16(*(const v8bf*)(pp), *(const v8bf*)(pp + 8));
#pragma unroll
      for (int j = 0; j < 4; ++j) {
        v16bf bv = *(const v16bf*)(&Vt[j * 16 + l16][ks * 32 + half * 16]);
        acc[j] = wmma_bf16(ap, bv, acc[j]);
      }
    }
  }

  // Epilogue: normalize and scatter to [b][s][h*64+d] bf16
  const int bb = bh >> 3, h = bh & 7;
#pragma unroll
  for (int r = 0; r < 8; ++r) {
    float inv = 1.f / rsum[r];
    int s = q0 + wave * 16 + half * 8 + r;
#pragma unroll
    for (int j = 0; j < 4; ++j) {
      int d = j * 16 + l16;
      long idx = ((long)(bb * S_ + s)) * INNER_ + h * DHEAD_ + d;
      O[idx] = (__bf16)(acc[j][r] * inv);
    }
  }
}

// ---------------------------------------------------------------------------
// Kernel 3: output projection. out[8192x512] = Aw @ Wout + b_out (fp32 out)
// ---------------------------------------------------------------------------
__global__ __launch_bounds__(256) void lsa_out_gemm(
    const __bf16* __restrict__ A,   // [8192][512]
    const float*  __restrict__ W,   // [512][512]
    const float*  __restrict__ bias,
    float* __restrict__ out) {
  __shared__ __bf16 Bt[64][512];    // 64 KB
  const int n0 = blockIdx.x * 64;
  const int m0 = blockIdx.y * 128;
  const int tid = threadIdx.x;

  for (int i = tid; i < 64 * 512; i += 256) {
    int n = i & 63, k = i >> 6;
    Bt[n][k] = (__bf16)W[(long)k * INNER_ + n0 + n];
  }
  __syncthreads();

  const int wave = tid >> 5, lane = tid & 31;
  const int l16 = lane & 15, half = (lane >> 4) & 1;
  const int arow = m0 + wave * 16 + l16;

  v8f acc[4] = {};
  for (int ks = 0; ks < 16; ++ks) {
    v16bf a = load_afrag(A, INNER_, arow, ks * 32, half);
#pragma unroll
    for (int j = 0; j < 4; ++j) {
      v16bf bk = load_bfrag(&Bt[0][0], 512, j * 16 + l16, ks * 32, half);
      acc[j] = wmma_bf16(a, bk, acc[j]);
    }
  }

#pragma unroll
  for (int j = 0; j < 4; ++j) {
#pragma unroll
    for (int r = 0; r < 8; ++r) {
      int m = m0 + wave * 16 + half * 8 + r;
      int n = n0 + j * 16 + l16;
      out[(long)m * DIM_ + n] = acc[j][r] + bias[n];
    }
  }
}

// ---------------------------------------------------------------------------
// Launcher
// ---------------------------------------------------------------------------
extern "C" void kernel_launch(void* const* d_in, const int* in_sizes, int n_in,
                              void* d_out, int out_size, void* d_ws, size_t ws_size,
                              hipStream_t stream) {
  const float* x    = (const float*)d_in[0];  // [4,2048,512]
  const float* Wqkv = (const float*)d_in[1];  // [512,1536]
  const float* logt = (const float*)d_in[2];  // scalar
  const float* Wout = (const float*)d_in[3];  // [512,512]
  const float* bout = (const float*)d_in[4];  // [512]
  float* out = (float*)d_out;

  char* ws = (char*)d_ws;
  const size_t nX   = (size_t)MROWS_ * DIM_;           // 4,194,304
  const size_t nQKV = (size_t)B_ * HEADS_ * S_ * DHEAD_; // 4,194,304 each
  __bf16* Xb = (__bf16*)ws;
  __bf16* Qw = Xb + nX;
  __bf16* Kw = Qw + nQKV;
  __bf16* Vw = Kw + nQKV;
  __bf16* Aw = Vw + nQKV;                              // attn out, nX elems

  lsa_cvt<<<(int)((nX + 255) / 256), 256, 0, stream>>>(x, Xb, (int)nX);

  lsa_qkv_gemm<<<dim3(NQKV_ / 64, MROWS_ / 128), 256, 0, stream>>>(
      Xb, Wqkv, logt, Qw, Kw, Vw);

  lsa_attn<<<dim3(S_ / 64, B_ * HEADS_), 128, 0, stream>>>(Qw, Kw, Vw, Aw);

  lsa_out_gemm<<<dim3(DIM_ / 64, MROWS_ / 128), 256, 0, stream>>>(
      Aw, Wout, bout, out);
}
